// NeuralODEHealthModel_64278480552282
// MI455X (gfx1250) — matrix operand order repaired and checked by reading, not compiled, not run on hardware
//
#include <hip/hip_runtime.h>
#include <hip/hip_bf16.h>

typedef __attribute__((ext_vector_type(16))) _Float16 v16h;
typedef __attribute__((ext_vector_type(8)))  _Float16 v8h;
typedef __attribute__((ext_vector_type(8)))  float    v8f;

#define ISC_REF 9.0f
#define ALPHA   0.0005f
#define T_REF   0.5f
#define NDAYS   365

// Branch-free transcendentals: lower to v_exp_f32 / v_rcp_f32, no EXEC masking.
__device__ __forceinline__ float sigf(float x) { return 1.0f / (1.0f + __expf(-x)); }
__device__ __forceinline__ float tanh_fast(float x) {
    // tanh(x) = 1 - 2/(exp(2x)+1); exp->inf => 1, exp->0 => -1 (overflow-safe)
    return 1.0f - 2.0f / (__expf(2.0f * x) + 1.0f);
}

// ---------------------------------------------------------------------------
// WMMA fragment helpers (wave32, V_WMMA_F32_16X16X32_F16 layouts: D per ISA 7.12.2, A/B contiguous in K)
// ---------------------------------------------------------------------------

// A (16x32, f16) from row-major LDS [16][ld]:
// lane l: M = l%16 ; K = kk + (l/16)*16 + 0..15 (contiguous in K).
__device__ __forceinline__ v16h frag_A_lds(const _Float16* sA, int ld, int kk) {
    int l  = threadIdx.x & 31;
    int m  = l & 15;
    int kh = (l >> 4) * 16;
    const _Float16* p = sA + m * ld + kk + kh;
    v8h lo = *(const v8h*)p;          // ds_load_b128
    v8h hi = *(const v8h*)(p + 8);    // ds_load_b128
    v16h r;
#pragma unroll
    for (int i = 0; i < 8; ++i) { r[i] = lo[i]; r[8 + i] = hi[i]; }
    return r;
}

// B (32x16, f16) gathered once from global row-major W[N][K] (B(k,n)=W[n][k]):
// lane l: N = n0 + l%16 ; K = k0 + (l/16)*16 + 0..15 (contiguous in K).
__device__ __forceinline__ v16h frag_B_gmem(const float* W, int ldk, int n0, int k0) {
    int l = threadIdx.x & 31;
    const float* p = W + (size_t)(n0 + (l & 15)) * ldk + k0 + ((l >> 4) << 4);
    v16h r;
#pragma unroll
    for (int e = 0; e < 16; ++e) r[e] = (_Float16)p[e];
    return r;
}

__device__ __forceinline__ v16h frag_B_gmem_pad(const float* W, int ldk, int Kv, int n0, int k0) {
    int l  = threadIdx.x & 31;
    int kb = k0 + ((l >> 4) << 4);
    const float* p = W + (size_t)(n0 + (l & 15)) * ldk;
    v16h r;
#pragma unroll
    for (int e = 0; e < 16; ++e) {
        int k = kb + e;
        r[e] = (k < Kv) ? (_Float16)p[k] : (_Float16)0.0f;
    }
    return r;
}

// D (16x16, f32): vgpr r, lane l -> m = r + 8*(l/16), n = l%16
__device__ __forceinline__ void store_D_f32(float* s, int ld, int n0, v8f acc) {
    int l  = threadIdx.x & 31;
    int n  = n0 + (l & 15);
    int mb = (l >> 4) * 8;
#pragma unroll
    for (int r = 0; r < 8; ++r) s[(mb + r) * ld + n] = acc[r];
}

__device__ __forceinline__ void store_relu_f16(_Float16* s, int ld, int n0, v8f acc, const float* bias) {
    int l  = threadIdx.x & 31;
    int n  = n0 + (l & 15);
    int mb = (l >> 4) * 8;
    float b = bias[n];
#pragma unroll
    for (int r = 0; r < 8; ++r) s[(mb + r) * ld + n] = (_Float16)fmaxf(acc[r] + b, 0.0f);
}

__device__ __forceinline__ void store_relu_f32(float* s, int ld, int n0, v8f acc, const float* bias) {
    int l  = threadIdx.x & 31;
    int n  = n0 + (l & 15);
    int mb = (l >> 4) * 8;
    float b = bias[n];
#pragma unroll
    for (int r = 0; r < 8; ++r) s[(mb + r) * ld + n] = fmaxf(acc[r] + b, 0.0f);
}

__device__ __forceinline__ v8f wmma_f16(v16h a, v16h b, v8f c) {
    return __builtin_amdgcn_wmma_f32_16x16x32_f16(false, a, false, b, (short)0, c, false, false);
}

// ---------------------------------------------------------------------------
// Kernel 1: RK45 (Dormand-Prince) neural-ODE trajectory. Tiny & serial: 1 block.
// traj[366][8] -> workspace
// ---------------------------------------------------------------------------
__global__ __launch_bounds__(64)
void ode_traj_kernel(const float* __restrict__ h0,
                     const float* __restrict__ W1, const float* __restrict__ b1,
                     const float* __restrict__ W2, const float* __restrict__ b2,
                     const float* __restrict__ W3, const float* __restrict__ b3,
                     float* __restrict__ traj) {
    __shared__ float h[8], ht[9], z1[64], z2[64], kb[6][8];
    const int tid = threadIdx.x;
    const float dt = 1.0f / (float)NDAYS;

    const float Ac[6][5] = {
        {0.f, 0.f, 0.f, 0.f, 0.f},
        {0.2f, 0.f, 0.f, 0.f, 0.f},
        {3.f/40.f, 9.f/40.f, 0.f, 0.f, 0.f},
        {44.f/45.f, -56.f/15.f, 32.f/9.f, 0.f, 0.f},
        {19372.f/6561.f, -25360.f/2187.f, 64448.f/6561.f, -212.f/729.f, 0.f},
        {9017.f/3168.f, -355.f/33.f, 46732.f/5247.f, 49.f/176.f, -5103.f/18656.f}};
    const float Cc[6] = {0.f, 0.2f, 0.3f, 0.8f, 8.f/9.f, 1.f};
    const float Bc[6] = {35.f/384.f, 0.f, 500.f/1113.f, 125.f/192.f, -2187.f/6784.f, 11.f/84.f};

    if (tid < 8) { h[tid] = h0[tid]; traj[tid] = h[tid]; }
    __syncthreads();

    for (int step = 0; step < NDAYS; ++step) {
        float t0 = (float)step / (float)NDAYS;
        for (int s = 0; s < 6; ++s) {
            if (tid < 8) {
                float a = h[tid];
                for (int j = 0; j < s; ++j) a += dt * Ac[s][j] * kb[j][tid];
                ht[tid] = a;
            }
            if (tid == 8) ht[8] = t0 + Cc[s] * dt;
            __syncthreads();
            {   // z1 = tanh(inp @ W1^T + b1), 64 units, K=9
                float a = b1[tid];
#pragma unroll
                for (int j = 0; j < 9; ++j) a += W1[tid * 9 + j] * ht[j];
                z1[tid] = tanh_fast(a);
            }
            __syncthreads();
            {   // z2 = tanh(z1 @ W2^T + b2), 64 units, K=64
                float a = b2[tid];
#pragma unroll
                for (int j = 0; j < 64; ++j) a += W2[tid * 64 + j] * z1[j];
                z2[tid] = tanh_fast(a);
            }
            __syncthreads();
            if (tid < 8) {   // k = z2 @ W3^T + b3, 8 units
                float a = b3[tid];
#pragma unroll
                for (int j = 0; j < 64; ++j) a += W3[tid * 64 + j] * z2[j];
                kb[s][tid] = a;
            }
            __syncthreads();
        }
        if (tid < 8) {
            float a = h[tid];
#pragma unroll
            for (int s = 0; s < 6; ++s) a += dt * Bc[s] * kb[s][tid];
            h[tid] = a;
            traj[(size_t)(step + 1) * 8 + tid] = a;
        }
        __syncthreads();
    }
}

// ---------------------------------------------------------------------------
// Kernel 2: fused 2-layer LSTM (WMMA f16) + fusion MLP (WMMA f16) + outputs.
// One block per 16 batch rows; 8 waves (256 threads); weights live in register
// B-fragments across all 2048 timesteps; activations staged through LDS.
// ---------------------------------------------------------------------------
__global__ __launch_bounds__(256, 1)
void fused_lstm_fusion_kernel(
    const float* __restrict__ X, const int* __restrict__ day_ids,
    const float* __restrict__ raw, const float* __restrict__ traj,
    const float* __restrict__ Wih0, const float* __restrict__ Whh0,
    const float* __restrict__ bih0, const float* __restrict__ bhh0,
    const float* __restrict__ Wih1, const float* __restrict__ Whh1,
    const float* __restrict__ bih1, const float* __restrict__ bhh1,
    const float* __restrict__ hdw, const float* __restrict__ hdb,
    const float* __restrict__ fW1, const float* __restrict__ fb1,
    const float* __restrict__ fW2, const float* __restrict__ fb2,
    const float* __restrict__ rW1, const float* __restrict__ rb1,
    const float* __restrict__ rW2, const float* __restrict__ rb2,
    float* __restrict__ out, int T, int B) {

    __shared__ __align__(16) _Float16 sA[16 * 128];    // A staging (all matmuls)
    __shared__ __align__(16) _Float16 sFa[16 * 64];    // feat1 (f16)
    __shared__ __align__(16) _Float16 sFbuf[16 * 64];  // feat2 (f16)
    __shared__ float sG[16 * 256];                     // gate accumulators
    __shared__ float sH0[16 * 64], sC0[16 * 64], sH1[16 * 64], sC1[16 * 64];
    __shared__ float sB0[256], sB1[256], sBf1[64], sBf2[64], sRb1v[32], sRW2v[32];
    __shared__ float sR[16 * 32];
    __shared__ float sHl[16 * 8], sD[16], sIp[16], sr[16], srb2;

    const int tid = threadIdx.x;
    const int wv  = __builtin_amdgcn_readfirstlane(tid >> 5);  // wave id, SGPR
    const int b0  = blockIdx.x * 16;
    const size_t BT = (size_t)B * (size_t)T;

    // ---- init LDS ----
#pragma unroll
    for (int ii = 0; ii < 4; ++ii) {
        int s = tid + 256 * ii;
        sH0[s] = 0.f; sC0[s] = 0.f; sH1[s] = 0.f; sC1[s] = 0.f;
    }
    sB0[tid] = bih0[tid] + bhh0[tid];
    sB1[tid] = bih1[tid] + bhh1[tid];
    if (tid < 64) { sBf1[tid] = fb1[tid]; sBf2[tid] = fb2[tid]; }
    if (tid < 32) { sRb1v[tid] = rb1[tid]; sRW2v[tid] = rW2[tid]; }
    if (tid == 0) srb2 = rb2[0];
    if (tid < 128) {
        int m = tid >> 3, j = tid & 7;
        sHl[tid] = traj[(size_t)day_ids[b0 + m] * 8 + j];
    }
    __syncthreads();
    if (tid < 16) {
        float a = hdb[0];
#pragma unroll
        for (int j = 0; j < 8; ++j) a += sHl[tid * 8 + j] * hdw[j];
        sD[tid] = sigf(a);
    }

    // ---- preload all weight B-fragments into registers (reused T times) ----
    v16h B0f[2][3], B1f[2][4];
#pragma unroll
    for (int tt = 0; tt < 2; ++tt) {
        int n0 = wv * 32 + tt * 16;
        B0f[tt][0] = frag_B_gmem(Wih0, 32, n0, 0);   // k 0..31  (x)
        B0f[tt][1] = frag_B_gmem(Whh0, 64, n0, 0);   // k 32..63 (h)
        B0f[tt][2] = frag_B_gmem(Whh0, 64, n0, 32);  // k 64..95 (h)
        B1f[tt][0] = frag_B_gmem(Wih1, 64, n0, 0);
        B1f[tt][1] = frag_B_gmem(Wih1, 64, n0, 32);
        B1f[tt][2] = frag_B_gmem(Whh1, 64, n0, 0);
        B1f[tt][3] = frag_B_gmem(Whh1, 64, n0, 32);
    }
    v16h BF1f[3], BF2f[2], BRf[2];
    if (wv < 4) {
#pragma unroll
        for (int kx = 0; kx < 3; ++kx) BF1f[kx] = frag_B_gmem_pad(fW1, 73, 73, wv * 16, kx * 32);
        BF2f[0] = frag_B_gmem(fW2, 64, wv * 16, 0);
        BF2f[1] = frag_B_gmem(fW2, 64, wv * 16, 32);
    } else if (wv < 6) {
        BRf[0] = frag_B_gmem(rW1, 64, (wv - 4) * 16, 0);
        BRf[1] = frag_B_gmem(rW1, 64, (wv - 4) * 16, 32);
    }

    // per-thread owned X element pair (row = tid/16, cols 2*(tid%16)..+1)
    const int xr = tid >> 4;
    const int xc = (tid & 15) * 2;
    const float* xptr = X + ((size_t)(b0 + xr) * T) * 32 + xc;
    float2 xreg = *(const float2*)(xptr);  // t = 0

    __syncthreads();

    for (int t = 0; t < T; ++t) {
        // ---- phase 1: stage A0 = [x_t | h0], physics current, prefetch next X
        sA[xr * 128 + xc]     = (_Float16)xreg.x;
        sA[xr * 128 + xc + 1] = (_Float16)xreg.y;
#pragma unroll
        for (int ii = 0; ii < 4; ++ii) {
            int s = tid + 256 * ii; int m = s >> 6, j = s & 63;
            sA[m * 128 + 32 + j] = (_Float16)sH0[m * 64 + j];
        }
        if (t + 1 < T) xreg = *(const float2*)(xptr + (size_t)(t + 1) * 32);
        if (t + 8 < T) __builtin_prefetch(xptr + (size_t)(t + 8) * 32, 0, 1);
        if (tid < 16) {
            const float* rp = raw + ((size_t)(b0 + tid) * T + t) * 4;
            sIp[tid] = rp[0] * ISC_REF * (1.0f + ALPHA * (rp[1] - T_REF));
        }
        __syncthreads();

        // ---- phase 2: layer0 gates = [x|h0] (16x96) @ [Wih0|Whh0]^T (96x256)
        {
            v16h a0 = frag_A_lds(sA, 128, 0);
            v16h a1 = frag_A_lds(sA, 128, 32);
            v16h a2 = frag_A_lds(sA, 128, 64);
#pragma unroll
            for (int tt = 0; tt < 2; ++tt) {
                v8f acc = {};
                acc = wmma_f16(a0, B0f[tt][0], acc);
                acc = wmma_f16(a1, B0f[tt][1], acc);
                acc = wmma_f16(a2, B0f[tt][2], acc);
                store_D_f32(sG, 256, wv * 32 + tt * 16, acc);
            }
        }
        __syncthreads();

        // ---- phase 3: layer0 cell; stage A1 = [h0_new | h1_old]
#pragma unroll
        for (int ii = 0; ii < 4; ++ii) {
            int s = tid + 256 * ii; int m = s >> 6, j = s & 63;
            float ig = sG[m * 256 + j]        + sB0[j];
            float fg = sG[m * 256 + 64 + j]   + sB0[64 + j];
            float gg = sG[m * 256 + 128 + j]  + sB0[128 + j];
            float og = sG[m * 256 + 192 + j]  + sB0[192 + j];
            float c  = sigf(fg) * sC0[m * 64 + j] + sigf(ig) * tanh_fast(gg);
            sC0[m * 64 + j] = c;
            float h = sigf(og) * tanh_fast(c);
            sH0[m * 64 + j] = h;
            sA[m * 128 + j]      = (_Float16)h;
            sA[m * 128 + 64 + j] = (_Float16)sH1[m * 64 + j];
        }
        __syncthreads();

        // ---- phase 4: layer1 gates = [env|h1] (16x128) @ [Wih1|Whh1]^T
        {
            v16h a0 = frag_A_lds(sA, 128, 0);
            v16h a1 = frag_A_lds(sA, 128, 32);
            v16h a2 = frag_A_lds(sA, 128, 64);
            v16h a3 = frag_A_lds(sA, 128, 96);
#pragma unroll
            for (int tt = 0; tt < 2; ++tt) {
                v8f acc = {};
                acc = wmma_f16(a0, B1f[tt][0], acc);
                acc = wmma_f16(a1, B1f[tt][1], acc);
                acc = wmma_f16(a2, B1f[tt][2], acc);
                acc = wmma_f16(a3, B1f[tt][3], acc);
                store_D_f32(sG, 256, wv * 32 + tt * 16, acc);
            }
        }
        __syncthreads();

        // ---- phase 5: layer1 cell; stage fusion A = [h1_new | health | Iph | 0]
#pragma unroll
        for (int ii = 0; ii < 4; ++ii) {
            int s = tid + 256 * ii; int m = s >> 6, j = s & 63;
            float ig = sG[m * 256 + j]        + sB1[j];
            float fg = sG[m * 256 + 64 + j]   + sB1[64 + j];
            float gg = sG[m * 256 + 128 + j]  + sB1[128 + j];
            float og = sG[m * 256 + 192 + j]  + sB1[192 + j];
            float c  = sigf(fg) * sC1[m * 64 + j] + sigf(ig) * tanh_fast(gg);
            sC1[m * 64 + j] = c;
            float h = sigf(og) * tanh_fast(c);
            sH1[m * 64 + j] = h;
            sA[m * 128 + j] = (_Float16)h;
        }
#pragma unroll
        for (int ii = 0; ii < 2; ++ii) {
            int s = tid + 256 * ii; int m = s >> 5, c = s & 31;  // col 64+c
            float v = (c < 8) ? sHl[m * 8 + c] : ((c == 8) ? sIp[m] : 0.0f);
            sA[m * 128 + 64 + c] = (_Float16)v;
        }
        __syncthreads();

        // ---- phase 6: feat1 = relu(z @ fus_W1^T + b1)  (16x64, K=96 padded)
        if (wv < 4) {
            v16h a0 = frag_A_lds(sA, 128, 0);
            v16h a1 = frag_A_lds(sA, 128, 32);
            v16h a2 = frag_A_lds(sA, 128, 64);
            v8f acc = {};
            acc = wmma_f16(a0, BF1f[0], acc);
            acc = wmma_f16(a1, BF1f[1], acc);
            acc = wmma_f16(a2, BF1f[2], acc);
            store_relu_f16(sFa, 64, wv * 16, acc, sBf1);
        }
        __syncthreads();

        // ---- phase 7: feat2 = relu(feat1 @ fus_W2^T + b2)  (16x64, K=64)
        if (wv < 4) {
            v16h a0 = frag_A_lds(sFa, 64, 0);
            v16h a1 = frag_A_lds(sFa, 64, 32);
            v8f acc = {};
            acc = wmma_f16(a0, BF2f[0], acc);
            acc = wmma_f16(a1, BF2f[1], acc);
            store_relu_f16(sFbuf, 64, wv * 16, acc, sBf2);
        }
        __syncthreads();

        // ---- phase 8: r1 = relu(feat2 @ res_W1^T + rb1)  (16x32, K=64)
        if (wv >= 4 && wv < 6) {
            v16h a0 = frag_A_lds(sFbuf, 64, 0);
            v16h a1 = frag_A_lds(sFbuf, 64, 32);
            v8f acc = {};
            acc = wmma_f16(a0, BRf[0], acc);
            acc = wmma_f16(a1, BRf[1], acc);
            store_relu_f32(sR, 32, (wv - 4) * 16, acc, sRb1v);
        }
        __syncthreads();

        // ---- phase 9: r = r1 @ res_W2^T + rb2  (N=1: VALU dot per row)
        if (tid < 16) {
            float a = srb2;
#pragma unroll
            for (int j = 0; j < 32; ++j) a += sR[tid * 32 + j] * sRW2v[j];
            sr[tid] = a;
        }
        __syncthreads();

        // ---- phase 10: write outputs (I_pred, I_phys, I_base, D, r, health_seq)
        if (tid < 208) {
            int row = tid & 15, f = tid >> 4;
            size_t tok = (size_t)(b0 + row) * T + t;
            float Iph = sIp[row], Dv = sD[row], rv = sr[row];
            if (f == 0)      out[tok]            = Dv * Iph + rv;
            else if (f == 1) out[BT + tok]       = Iph;
            else if (f == 2) out[2 * BT + tok]   = Dv * Iph;
            else if (f == 3) out[3 * BT + tok]   = Dv;
            else if (f == 4) out[4 * BT + tok]   = rv;
            else             out[5 * BT + tok * 8 + (size_t)(f - 5)] = sHl[row * 8 + (f - 5)];
        }
        __syncthreads();  // protect sIp/sr before next iteration's phase 1
    }
}

// ---------------------------------------------------------------------------
extern "C" void kernel_launch(void* const* d_in, const int* in_sizes, int n_in,
                              void* d_out, int out_size, void* d_ws, size_t ws_size,
                              hipStream_t stream) {
    const float* X    = (const float*)d_in[0];
    const int*   dids = (const int*)d_in[1];
    const float* raw  = (const float*)d_in[2];
    const float* ih0  = (const float*)d_in[3];
    const float* oW1  = (const float*)d_in[4];
    const float* ob1  = (const float*)d_in[5];
    const float* oW2  = (const float*)d_in[6];
    const float* ob2  = (const float*)d_in[7];
    const float* oW3  = (const float*)d_in[8];
    const float* ob3  = (const float*)d_in[9];
    const float* Wih0 = (const float*)d_in[10];
    const float* Whh0 = (const float*)d_in[11];
    const float* bih0 = (const float*)d_in[12];
    const float* bhh0 = (const float*)d_in[13];
    const float* Wih1 = (const float*)d_in[14];
    const float* Whh1 = (const float*)d_in[15];
    const float* bih1 = (const float*)d_in[16];
    const float* bhh1 = (const float*)d_in[17];
    const float* hdw  = (const float*)d_in[18];
    const float* hdb  = (const float*)d_in[19];
    const float* fW1  = (const float*)d_in[20];
    const float* fb1  = (const float*)d_in[21];
    const float* fW2  = (const float*)d_in[22];
    const float* fb2  = (const float*)d_in[23];
    const float* rW1  = (const float*)d_in[24];
    const float* rb1  = (const float*)d_in[25];
    const float* rW2  = (const float*)d_in[26];
    const float* rb2  = (const float*)d_in[27];

    const int B = in_sizes[1];                 // 256
    const int T = in_sizes[0] / (B * 32);      // 2048
    float* traj = (float*)d_ws;                // [366][8]
    float* out  = (float*)d_out;

    ode_traj_kernel<<<1, 64, 0, stream>>>(ih0, oW1, ob1, oW2, ob2, oW3, ob3, traj);
    fused_lstm_fusion_kernel<<<B / 16, 256, 0, stream>>>(
        X, dids, raw, traj,
        Wih0, Whh0, bih0, bhh0, Wih1, Whh1, bih1, bhh1,
        hdw, hdb, fW1, fb1, fW2, fb2, rW1, rb1, rW2, rb2,
        out, T, B);
}
